// MoEUNetWrapper_11269994185012
// MI455X (gfx1250) — compile-verified, matrix operand order checked by reading, not run on hardware
//
#include <hip/hip_runtime.h>
#include <stdint.h>

// Problem constants (match reference)
#define BATCH 16
#define CIN   320
#define COUT  320
#define HH    64
#define WW    64
#define NCLS  5
#define HW    (HH * WW)          // 4096
#define PH    (HH + 2)           // 66 padded
#define PW    (WW + 2)           // 66 padded
#define PHW   (PH * PW)          // 4356

// Tiling
#define MT 64      // workgroup M tile (COUT)
#define NT 256     // workgroup N tile (pixels) = 4 image rows of 64
#define KT 32      // K chunk (channels) per WMMA group
#define KCHUNKS (CIN / KT)       // 10
#define NITER  (9 * KCHUNKS)     // 90
#define NWAVES 8   // 256 threads, wave32

typedef __attribute__((ext_vector_type(16))) __bf16    v16bf;
typedef __attribute__((ext_vector_type(8)))  float     v8f;
typedef __attribute__((ext_vector_type(4)))  uint32_t  u32x4;
typedef __attribute__((ext_vector_type(8)))  uint32_t  u32x8;

#ifndef HAVE_TDM
#if defined(__has_builtin)
#if __has_builtin(__builtin_amdgcn_tensor_load_to_lds)
#define HAVE_TDM 1
#else
#define HAVE_TDM 0
#endif
#else
#define HAVE_TDM 0
#endif
#endif

typedef __attribute__((ext_vector_type(4))) uint32_t tdm_u32x4;
typedef __attribute__((ext_vector_type(8))) int      tdm_i32x8;
typedef __attribute__((ext_vector_type(4))) int      tdm_i32x4;

__device__ __forceinline__ uint16_t f32_to_bf16(float f) {
  uint32_t u = __builtin_bit_cast(uint32_t, f);
  uint32_t r = u + 0x7FFFu + ((u >> 16) & 1u);
  return (uint16_t)(r >> 16);
}

// Assumed per-lane 16-bit fragment layout (CDNA5 ISA 7.12.2, 16x32):
//   row = lane%16, kbase = (lane/16)*8
//   VGPR0..3 hold K=kbase..kbase+7, VGPR4..7 hold K=kbase+16..kbase+23
__device__ __forceinline__ v16bf load_frag(const uint16_t* row_base, int kbase) {
  u32x4 lo = *(const u32x4*)(row_base + kbase);
  u32x4 hi = *(const u32x4*)(row_base + kbase + 16);
  u32x8 c;
  c[0] = lo[0]; c[1] = lo[1]; c[2] = lo[2]; c[3] = lo[3];
  c[4] = hi[0]; c[5] = hi[1]; c[6] = hi[2]; c[7] = hi[3];
  return __builtin_bit_cast(v16bf, c);
}

#if HAVE_TDM
// Fire a TDM tile load of bf16 data (data_size = 2B).
// Tile: tile_d0 (contig) x tile_d1 (stride d0_stride) x tile_d2 (stride
// d1_stride); tile_d2 == 0 -> 2D tile. LDS receives the tile densely packed
// in [d2][d1][d0] order. D# field packing per CDNA5 ISA 8.3-8.5.
__device__ __forceinline__ void tdm_load_to_lds(uint32_t lds_off, uint64_t gaddr,
                                                uint32_t tile_d0, uint32_t tile_d1,
                                                uint32_t tile_d2,
                                                uint64_t d0_stride, uint64_t d1_stride,
                                                uint32_t tensor_d0, uint32_t tensor_d1,
                                                uint32_t tensor_d2) {
  tdm_u32x4 g0;
  g0[0] = 1u;                                            // count=1, user descriptor
  g0[1] = lds_off;                                       // lds_addr
  g0[2] = (uint32_t)(gaddr & 0xFFFFFFFFu);               // global_addr[31:0]
  g0[3] = (uint32_t)((gaddr >> 32) & 0x01FFFFFFu)        // global_addr[56:32]
          | (2u << 30);                                  // type=2 ("image")
  tdm_i32x8 g1;
  g1[0] = (int)(1u << 16);                               // data_size=1 (2 bytes)
  g1[1] = (int)((tensor_d0 & 0xFFFFu) << 16);            // tensor_dim0[15:0]
  g1[2] = (int)((tensor_d0 >> 16) | ((tensor_d1 & 0xFFFFu) << 16));
  g1[3] = (int)((tensor_d1 >> 16) | (tile_d0 << 16));    // tile_dim0
  g1[4] = (int)(tile_d1 | (tile_d2 << 16));              // tile_dim1, tile_dim2
  g1[5] = (int)(uint32_t)(d0_stride & 0xFFFFFFFFu);      // tensor_dim0_stride lo
  g1[6] = (int)(uint32_t)(((d0_stride >> 32) & 0xFFFFu)  // stride0 hi
                          | ((d1_stride & 0xFFFFu) << 16));  // tensor_dim1_stride lo
  g1[7] = (int)(uint32_t)((d1_stride >> 16) & 0xFFFFFFFFu);  // stride1 hi
  tdm_i32x4 g2;
  g2[0] = (int)tensor_d2;                                // tensor_dim2
  g2[1] = 0;                                             // tensor_dim3
  g2[2] = 0;                                             // tensor_dim2_stride lo
  g2[3] = 0;                                             // stride hi / tile_dim3
  tdm_i32x4 z4 = {0, 0, 0, 0};
#if __clang_major__ >= 23
  tdm_i32x8 z8 = {0, 0, 0, 0, 0, 0, 0, 0};
  __builtin_amdgcn_tensor_load_to_lds(g0, g1, g2, z4, z8, 0);
#else
  __builtin_amdgcn_tensor_load_to_lds(g0, g1, g2, z4, 0);
#endif
}
#endif  // HAVE_TDM

// ---------------------------------------------------------------------------
// One-time conversion kernels (memory-bound, ~6us total at 23.3 TB/s)
// ---------------------------------------------------------------------------

// features f32 [B][CIN][H][W] -> bf16 channel-last PADDED:
// Xp[b][py*66+px][ci], zero border (py,px==0 or 65). 4 ci per thread.
__global__ __launch_bounds__(256)
void cvt_features_pad(const float* __restrict__ X, uint16_t* __restrict__ O, int n4) {
  int gid = blockIdx.x * blockDim.x + threadIdx.x;
  if (gid >= n4) return;
  const int ci   = (gid % (CIN / 4)) * 4;
  const int rest = gid / (CIN / 4);
  const int pix  = rest % PHW;
  const int b    = rest / PHW;
  const int py = pix / PW;
  const int px = pix % PW;
  uint2 o = make_uint2(0u, 0u);
  if (py >= 1 && py <= HH && px >= 1 && px <= WW) {
    const float* src = X + ((size_t)b * CIN + ci) * HW + (py - 1) * WW + (px - 1);
    o.x = (uint32_t)f32_to_bf16(src[0 * HW]) | ((uint32_t)f32_to_bf16(src[1 * HW]) << 16);
    o.y = (uint32_t)f32_to_bf16(src[2 * HW]) | ((uint32_t)f32_to_bf16(src[3 * HW]) << 16);
  }
  ((uint2*)O)[gid] = o;
}

// weights f32 [c][co][ci][kh][kw] -> bf16 [c][tap][co][ci]  (tap = kh*3+kw)
__global__ __launch_bounds__(256)
void cvt_weights(const float* __restrict__ Wt, uint16_t* __restrict__ O, int n) {
  int o = blockIdx.x * blockDim.x + threadIdx.x;
  if (o < n) {
    int ci  = o % CIN;
    int t1  = o / CIN;
    int co  = t1 % COUT;
    int t2  = t1 / COUT;
    int tap = t2 % 9;
    int c   = t2 / 9;
    O[o] = f32_to_bf16(Wt[(((size_t)c * COUT + co) * CIN + ci) * 9 + tap]);
  }
}

// ---------------------------------------------------------------------------
// Main GEMM kernel: all tile movement through the Tensor Data Mover,
// double-buffered on TENSORcnt; hot loop = ds_load_b128 frags + 8 WMMAs.
// ---------------------------------------------------------------------------
__global__ __launch_bounds__(256)
void moe_conv_wmma_tdm(const uint16_t* __restrict__ Xp,   // [B][66*66][CIN] bf16
                       const uint16_t* __restrict__ Wb,   // [NCLS][9][COUT][CIN] bf16
                       const float* __restrict__ Bias,    // [NCLS][COUT]
                       const int*   __restrict__ cls,     // [B]
                       float*       __restrict__ Y) {     // [B][COUT][HW]
  __shared__ __align__(16) uint16_t As[2][MT * KT];  // 2 x  4 KB: [co][k]
  __shared__ __align__(16) uint16_t Bs[2][NT * KT];  // 2 x 16 KB: [pix][k]

  const int ntiles = HW / NT;                        // 16
  const int mtiles = COUT / MT;                      // 5
  const int blk = blockIdx.x;
  const int nt = blk % ntiles;
  const int mt = (blk / ntiles) % mtiles;
  const int b  = blk / (ntiles * mtiles);

  const int c  = cls[b];
  const int h0 = nt * 4;                             // first output row of tile

  const int tid  = threadIdx.x;
  const int wave = tid >> 5;
  const int lane = tid & 31;
  const int lrow  = lane & 15;
  const int kbase = (lane >> 4) * 8;
  const int nb = wave * 32;                          // wave's N offset in tile

  const uint16_t* Xb = Xp + (size_t)b * PHW * CIN;

  v8f acc[4][2] = {};

  // Issue the (TDM or manual) tile loads for flat iteration `it` into buffer it&1.
  auto issue = [&](int it) {
    const int tap = it / KCHUNKS;
    const int kc  = it % KCHUNKS;
    const int buf = it & 1;
    const int dh = tap / 3 - 1;
    const int dw = tap % 3 - 1;
    // A tile: Wb[c][tap][mt*MT + 0..63][kc*KT + 0..31], row stride CIN
    const size_t w_elem = (((size_t)c * 9 + tap) * COUT + mt * MT) * CIN + kc * KT;
    // B tile: padded channel-last, rows h0+dh+1 .. +3, cols dw+1 .. +63 (in-bounds)
    const size_t x_elem = ((size_t)(h0 + dh + 1) * PW + (dw + 1)) * CIN + kc * KT;
#if HAVE_TDM
    if (wave == 0) {
      tdm_load_to_lds((uint32_t)(uintptr_t)(void*)&As[buf][0],
                      (uint64_t)(uintptr_t)(Wb + w_elem),
                      /*tile_d0=*/KT, /*tile_d1=*/MT, /*tile_d2=*/0,
                      /*d0_stride=*/CIN, /*d1_stride=*/0,
                      /*tensor_d0=*/CIN, /*tensor_d1=*/COUT, /*tensor_d2=*/0);
      tdm_load_to_lds((uint32_t)(uintptr_t)(void*)&Bs[buf][0],
                      (uint64_t)(uintptr_t)(Xb + x_elem),
                      /*tile_d0=*/KT, /*tile_d1=*/WW, /*tile_d2=*/4,
                      /*d0_stride=*/CIN, /*d1_stride=*/(uint64_t)PW * CIN,
                      /*tensor_d0=*/CIN, /*tensor_d1=*/PW, /*tensor_d2=*/PH);
    }
#else
    // Manual staging: A = one b128 per thread, B = four b128 per thread.
    {
      const int a_row = tid >> 2, a_chunk = tid & 3;
      *(u32x4*)&As[buf][a_row * KT + a_chunk * 8] =
          *(const u32x4*)(Wb + w_elem + (size_t)a_row * CIN + a_chunk * 8);
      const uint16_t* brow = Xb + x_elem + ((size_t)(tid >> 6) * PW + (tid & 63)) * CIN;
      u32x4* dst = (u32x4*)&Bs[buf][tid * KT];
      #pragma unroll
      for (int q = 0; q < 4; ++q) dst[q] = ((const u32x4*)brow)[q];
    }
#endif
  };

  // Prologue: fill buffer 0.
  issue(0);
#if HAVE_TDM
  if (wave == 0) __builtin_amdgcn_s_wait_tensorcnt(0);
#endif
  __syncthreads();

  for (int it = 0; it < NITER; ++it) {
    const int buf = it & 1;

    // Kick DMA for the next chunk into the other buffer (overlaps compute).
    if (it + 1 < NITER) issue(it + 1);

    // ---- Fragments + 8 WMMAs from current buffer ----
    v16bf afrag[4];
    #pragma unroll
    for (int m = 0; m < 4; ++m)
      afrag[m] = load_frag(&As[buf][(m * 16 + lrow) * KT], kbase);

    v16bf bfrag[2];
    #pragma unroll
    for (int j = 0; j < 2; ++j)
      bfrag[j] = load_frag(&Bs[buf][(nb + j * 16 + lrow) * KT], kbase);

    #pragma unroll
    for (int m = 0; m < 4; ++m)
      #pragma unroll
      for (int j = 0; j < 2; ++j)
        acc[m][j] = __builtin_amdgcn_wmma_f32_16x16x32_bf16(
            false, afrag[m], false, bfrag[j],
            (short)0, acc[m][j], false, false);

#if HAVE_TDM
    if (wave == 0) __builtin_amdgcn_s_wait_tensorcnt(0);  // next tiles landed
#endif
    __syncthreads();
  }

  // ---- Epilogue: bias + store (documented 16x16 f32 C/D layout) ----
  const int hiM = (lane >> 4) * 8;
  #pragma unroll
  for (int m = 0; m < 4; ++m) {
    #pragma unroll
    for (int j = 0; j < 2; ++j) {
      #pragma unroll
      for (int r = 0; r < 8; ++r) {
        const int co = mt * MT + m * 16 + r + hiM;
        const int p  = nt * NT + nb + j * 16 + lrow;
        Y[(((size_t)b * COUT + co) << 12) + p] = acc[m][j][r] + Bias[c * COUT + co];
      }
    }
  }
}

// ---------------------------------------------------------------------------
// Fallback: f32 global operands, converts in-kernel (used only if the
// workspace is too small for the bf16 staging buffers).
// ---------------------------------------------------------------------------
__global__ __launch_bounds__(256)
void moe_conv_wmma_f32(const float* __restrict__ X, const float* __restrict__ Wt,
                       const float* __restrict__ Bias, const int* __restrict__ cls,
                       float* __restrict__ Y) {
  __shared__ __align__(16) uint16_t As[MT * KT];
  __shared__ __align__(16) uint16_t Bs[NT * KT];

  const int ntiles = HW / NT, mtiles = COUT / MT;
  const int blk = blockIdx.x;
  const int nt = blk % ntiles;
  const int mt = (blk / ntiles) % mtiles;
  const int b  = blk / (ntiles * mtiles);
  const int c  = cls[b];

  const int tid = threadIdx.x, wave = tid >> 5, lane = tid & 31;
  const int lrow = lane & 15, kbase = (lane >> 4) * 8, nb = wave * 32;

  v8f acc[4][2] = {};
  const int p_stage = nt * NT + tid;
  const int h_stage = p_stage >> 6, w_stage = p_stage & 63;
  const float* Xb = X + (size_t)b * CIN * HW;
  const float* Wc = Wt + (size_t)c * COUT * CIN * 9;

  for (int tap = 0; tap < 9; ++tap) {
    const int dh = tap / 3 - 1, dw = tap % 3 - 1;
    const int hh2 = h_stage + dh, ww2 = w_stage + dw;
    const bool inb = ((unsigned)hh2 < (unsigned)HH) && ((unsigned)ww2 < (unsigned)WW);
    const int xoff = hh2 * WW + ww2;

    for (int kc = 0; kc < KCHUNKS; ++kc) {
      const int ci0 = kc * KT;
      {
        const int kk = lane;
        #pragma unroll
        for (int e = 0; e < 8; ++e) {
          const int co_l = (tid >> 5) + NWAVES * e;
          As[co_l * KT + kk] =
              f32_to_bf16(Wc[((mt * MT + co_l) * CIN + (ci0 + kk)) * 9 + tap]);
        }
      }
      {
        const float* xp = Xb + (size_t)ci0 * HW + xoff;
        uint32_t pk[16];
        #pragma unroll
        for (int k2 = 0; k2 < 16; ++k2) {
          float f0 = inb ? xp[(2 * k2 + 0) * HW] : 0.0f;
          float f1 = inb ? xp[(2 * k2 + 1) * HW] : 0.0f;
          pk[k2] = (uint32_t)f32_to_bf16(f0) | ((uint32_t)f32_to_bf16(f1) << 16);
        }
        u32x4* dst = (u32x4*)&Bs[tid * KT];
        #pragma unroll
        for (int q = 0; q < 4; ++q) {
          u32x4 v;
          v[0] = pk[4 * q + 0]; v[1] = pk[4 * q + 1];
          v[2] = pk[4 * q + 2]; v[3] = pk[4 * q + 3];
          dst[q] = v;
        }
      }
      __syncthreads();
      v16bf afrag[4];
      #pragma unroll
      for (int m = 0; m < 4; ++m)
        afrag[m] = load_frag(&As[(m * 16 + lrow) * KT], kbase);
      v16bf bfrag[2];
      #pragma unroll
      for (int j = 0; j < 2; ++j)
        bfrag[j] = load_frag(&Bs[(nb + j * 16 + lrow) * KT], kbase);
      #pragma unroll
      for (int m = 0; m < 4; ++m)
        #pragma unroll
        for (int j = 0; j < 2; ++j)
          acc[m][j] = __builtin_amdgcn_wmma_f32_16x16x32_bf16(
              false, afrag[m], false, bfrag[j], (short)0, acc[m][j], false, false);
      __syncthreads();
    }
  }
  const int hiM = (lane >> 4) * 8;
  #pragma unroll
  for (int m = 0; m < 4; ++m)
    #pragma unroll
    for (int j = 0; j < 2; ++j)
      #pragma unroll
      for (int r = 0; r < 8; ++r) {
        const int co = mt * MT + m * 16 + r + hiM;
        const int p  = nt * NT + nb + j * 16 + lrow;
        Y[(((size_t)b * COUT + co) << 12) + p] = acc[m][j][r] + Bias[c * COUT + co];
      }
}

extern "C" void kernel_launch(void* const* d_in, const int* in_sizes, int n_in,
                              void* d_out, int out_size, void* d_ws, size_t ws_size,
                              hipStream_t stream) {
  const float* X    = (const float*)d_in[0];   // [16,320,64,64] f32
  const float* Wt   = (const float*)d_in[1];   // [5,320,320,3,3] f32
  const float* Bias = (const float*)d_in[2];   // [5,320] f32
  const int*   cls  = (const int*)d_in[3];     // [16]
  float*       Y    = (float*)d_out;

  const int mtiles = COUT / MT;                // 5
  const int ntiles = HW / NT;                  // 16
  dim3 gemm_grid(BATCH * mtiles * ntiles);     // 1280 workgroups
  dim3 block(256);

  const size_t featB   = (size_t)BATCH * PHW * CIN * 2;       // 44.6 MB bf16 padded
  const size_t weightB = (size_t)NCLS * 9 * COUT * CIN * 2;   //  9.2 MB bf16

  if (ws_size >= featB + weightB) {
    uint16_t* Xp = (uint16_t*)d_ws;
    uint16_t* Wb = (uint16_t*)((char*)d_ws + featB);

    const int nfeat4 = BATCH * PHW * CIN / 4;
    cvt_features_pad<<<(nfeat4 + 255) / 256, 256, 0, stream>>>(X, Xp, nfeat4);
    const int nw = NCLS * 9 * COUT * CIN;
    cvt_weights<<<(nw + 255) / 256, 256, 0, stream>>>(Wt, Wb, nw);

    moe_conv_wmma_tdm<<<gemm_grid, block, 0, stream>>>(Xp, Wb, Bias, cls, Y);
  } else {
    moe_conv_wmma_f32<<<gemm_grid, block, 0, stream>>>(X, Wt, Bias, cls, Y);
  }
}
